// MaskedAttention_81561428951531
// MI455X (gfx1250) — matrix operand order repaired and checked by reading, not compiled, not run on hardware
//
#include <hip/hip_runtime.h>
#include <hip/hip_bf16.h>

typedef __attribute__((ext_vector_type(16))) __bf16       v16bf;
typedef __attribute__((ext_vector_type(8)))  __bf16       bf16x8;
typedef __attribute__((ext_vector_type(8)))  float        v8f;
typedef __attribute__((ext_vector_type(4)))  unsigned int v4u;
typedef __attribute__((ext_vector_type(8)))  int          v8i;
typedef __attribute__((ext_vector_type(4)))  int          v4i;

#define B_     2
#define N_     2048
#define D_     1024
#define H_     16
#define HD_    64
#define INNER_ 1024
#define NC3    (3 * INNER_)   /* 3072 */
#define ROWS   (B_ * N_)      /* 4096 */
#define EPS_   1e-5f
#define SCALE_ 0.125f         /* HD^-0.5 */

#if defined(__gfx1250__) && __has_builtin(__builtin_amdgcn_tensor_load_to_lds) && \
    __has_builtin(__builtin_amdgcn_s_wait_tensorcnt)
#define USE_TDM 1
#endif

__device__ __forceinline__ v8f wmma_bf16(v16bf a, v16bf b, v8f c) {
  // D = A(16x32 bf16) x B(32x16 bf16) + C(16x16 f32)
  return __builtin_amdgcn_wmma_f32_16x16x32_bf16(false, a, false, b, (short)0, c,
                                                 false, false);
}

// ---------------------------------------------------------------------------
// Kernel 1: LayerNorm, f32 -> bf16 normalized activations
// ---------------------------------------------------------------------------
__global__ void __launch_bounds__(256)
ln_kernel(const float* __restrict__ x, const float* __restrict__ gamma,
          const float* __restrict__ beta, __bf16* __restrict__ xn) {
  const int row = blockIdx.x;
  const float* xr = x + (size_t)row * D_;
  const int tid = threadIdx.x;

  float v[4];
  float s = 0.f, s2 = 0.f;
#pragma unroll
  for (int j = 0; j < 4; ++j) {
    v[j] = xr[tid + 256 * j];
    s += v[j];
    s2 += v[j] * v[j];
  }
#pragma unroll
  for (int d = 1; d < 32; d <<= 1) {
    s  += __shfl_xor(s, d, 32);
    s2 += __shfl_xor(s2, d, 32);
  }
  __shared__ float ss[8], ss2[8];
  const int wave = tid >> 5, lane = tid & 31;
  if (lane == 0) { ss[wave] = s; ss2[wave] = s2; }
  __syncthreads();
  if (tid == 0) {
    float t = 0.f, t2 = 0.f;
#pragma unroll
    for (int w = 0; w < 8; ++w) { t += ss[w]; t2 += ss2[w]; }
    ss[0] = t; ss2[0] = t2;
  }
  __syncthreads();
  const float mu   = ss[0] * (1.f / D_);
  const float var  = ss2[0] * (1.f / D_) - mu * mu;
  const float rstd = rsqrtf(var + EPS_);
#pragma unroll
  for (int j = 0; j < 4; ++j) {
    const int c = tid + 256 * j;
    xn[(size_t)row * D_ + c] = (__bf16)((v[j] - mu) * rstd * gamma[c] + beta[c]);
  }
}

// ---------------------------------------------------------------------------
// Kernel 2: f32 -> bf16 weight conversion (grid-stride)
// ---------------------------------------------------------------------------
__global__ void cvt_kernel(const float* __restrict__ src, __bf16* __restrict__ dst,
                           int n) {
  int i = blockIdx.x * blockDim.x + threadIdx.x;
  const int stride = gridDim.x * blockDim.x;
  for (; i < n; i += stride) dst[i] = (__bf16)src[i];
}

// ---------------------------------------------------------------------------
// Kernel 3: QKV GEMM (bf16 WMMA) + RoPE epilogue.
// B tiles staged TRANSPOSED (Bt[n][k]) so every WMMA B-fragment is two
// contiguous ds_load_b128 per lane (no scalar gather / v_mov packing).
// ---------------------------------------------------------------------------
__global__ void __launch_bounds__(128)
gemm_qkv_kernel(const __bf16* __restrict__ xn, const __bf16* __restrict__ wq,
                __bf16* __restrict__ qg, __bf16* __restrict__ kg,
                __bf16* __restrict__ vg) {
  __shared__ __bf16 As[64][32];
  __shared__ __bf16 Bt[64][32];   // transposed: Bt[n][k]
  const int c0 = blockIdx.x * 64;
  const int row0 = blockIdx.y * 64;
  const int tid = threadIdx.x, lane = tid & 31, wave = tid >> 5;
  const int hi = lane >> 4, lm = lane & 15;

  const v8f vzero = {};
  v8f acc[4];
#pragma unroll
  for (int i = 0; i < 4; ++i) acc[i] = vzero;

  for (int k0 = 0; k0 < D_; k0 += 32) {
#pragma unroll
    for (int q = 0; q < 2; ++q) {                 // A tile 64x32 (vector)
      const int ch = tid + 128 * q;
      const int r = ch >> 2, c8 = (ch & 3) * 8;
      *(bf16x8*)&As[r][c8] =
          *(const bf16x8*)&xn[(size_t)(row0 + r) * D_ + k0 + c8];
    }
#pragma unroll
    for (int q = 0; q < 2; ++q) {                 // B tile 32x64 -> Bt[64][32]
      const int ch = tid + 128 * q;
      const int r = ch >> 3, c8 = (ch & 7) * 8;   // r = k row, c8 = col
      const bf16x8 w = *(const bf16x8*)&wq[(size_t)(k0 + r) * NC3 + c0 + c8];
#pragma unroll
      for (int e = 0; e < 8; ++e) Bt[c8 + e][r] = w[e];
    }
    if (k0 + 32 < D_)
      __builtin_prefetch(&xn[(size_t)row0 * D_ + k0 + 32], 0, 1);
    __syncthreads();

    v16bf a;                                      // A frag: contiguous 16
#pragma unroll
    for (int e = 0; e < 16; ++e) {
      const int kk = 16 * hi + e;
      a[e] = As[wave * 16 + lm][kk];
    }
#pragma unroll
    for (int ct = 0; ct < 4; ++ct) {
      v16bf bb;                                   // B frag: contiguous 16
#pragma unroll
      for (int e = 0; e < 16; ++e) bb[e] = Bt[ct * 16 + lm][16 * hi + e];
      acc[ct] = wmma_bf16(a, bb, acc[ct]);
    }
    __syncthreads();
  }

  const int part = c0 / INNER_;             // 0=q 1=k 2=v
  const int h = (c0 % INNER_) / HD_;
#pragma unroll
  for (int r = 0; r < 8; ++r) {
    const int grow = row0 + wave * 16 + r + 8 * hi;   // D-layout row
    const int b = grow >> 11;                          // /N_
    const int pos = grow & (N_ - 1);
    const size_t base = ((size_t)(b * H_ + h) * N_ + pos) * HD_;
    if (part == 2) {
#pragma unroll
      for (int ct = 0; ct < 4; ++ct)
        vg[base + ct * 16 + lm] = (__bf16)acc[ct][r];
    } else {
      __bf16* dst = (part == 0) ? qg : kg;
#pragma unroll
      for (int ct = 0; ct < 2; ++ct) {
        const int d = ct * 16 + lm;                    // 0..31
        const float ifq = __powf(10000.f, -(float)d * (1.f / 32.f));
        const float ang = (float)pos * ifq;
        float sn, cs;
        __sincosf(ang, &sn, &cs);
        const float t1 = acc[ct][r], t2 = acc[ct + 2][r];
        dst[base + d]      = (__bf16)(t1 * cs - t2 * sn);
        dst[base + d + 32] = (__bf16)(t2 * cs + t1 * sn);
      }
    }
  }
}

// ---------------------------------------------------------------------------
// Kernel 4: flash attention. 1 wave per (b, h, 16-row i-tile); j-tiles of 32.
// Mask tile loaded via the Tensor Data Mover (D# built per ISA 08 §8.3/8.4),
// V tile staged transposed so P@V B-frags are contiguous ds_load_b128.
// ---------------------------------------------------------------------------
__global__ void __launch_bounds__(32)
flash_kernel(const __bf16* __restrict__ qg, const __bf16* __restrict__ kg,
             const __bf16* __restrict__ vg, const int* __restrict__ mask,
             __bf16* __restrict__ og) {
  const int it = blockIdx.x, h = blockIdx.y, b = blockIdx.z;
  const int i0 = it * 16;
  const int lane = threadIdx.x & 31, hi = lane >> 4, lm = lane & 15;
  const size_t bh = (size_t)(b * H_ + h) * N_;

  // Single shared struct => known layout; Ms at LDS segment offset 0 for TDM.
  __shared__ struct __align__(16) {
    int    Ms[16][32];    // 2048 B @ 0
    __bf16 Ps[16][32];    // 1024 B @ 2048
    __bf16 Vt[64][32];    // 4096 B @ 3072   (transposed: Vt[d][j_local])
  } sm;

  v16bf aq[2];                                    // Q frags, K-dim 0..31 / 32..63
#pragma unroll
  for (int c = 0; c < 2; ++c) {
    const __bf16* qp = qg + (bh + i0 + lm) * HD_ + c * 32 + 8 * hi;
    const bf16x8 lo = *(const bf16x8*)qp;
    const bf16x8 hi8 = *(const bf16x8*)(qp + 16);
#pragma unroll
    for (int e = 0; e < 8; ++e) { aq[c][e] = lo[e]; aq[c][8 + e] = hi8[e]; }
  }

  float mrow[8], lrow[8];
#pragma unroll
  for (int r = 0; r < 8; ++r) { mrow[r] = -1e30f; lrow[r] = 0.f; }
  const v8f vzero = {};
  v8f acc[4];
#pragma unroll
  for (int i = 0; i < 4; ++i) acc[i] = vzero;

  for (int j0 = 0; j0 < N_; j0 += 32) {
#ifdef USE_TDM
    {
      // Tensor DMA: 2D tile, 32(x) x 16(y) int32, row stride N_, into Ms @ 0.
      const unsigned long long ga =
          (unsigned long long)&mask[((size_t)b * N_ + i0) * N_ + j0];
      const v4u g0 = { 1u,                                   // count=1 (valid)
                       0u,                                   // lds_addr = 0 (Ms)
                       (unsigned int)(ga & 0xFFFFFFFFu),
                       (unsigned int)((ga >> 32) & 0x01FFFFFFu) | (2u << 30) };
      const v8i g1 = { (int)(2u << 16),                      // data_size=4B
                       (int)(((unsigned)N_ & 0xFFFFu) << 16),// tensor_dim0 lo16
                       (int)((((unsigned)N_ >> 16) & 0xFFFFu) |
                             (((unsigned)N_ & 0xFFFFu) << 16)), // dim0 hi|dim1 lo
                       (int)((((unsigned)N_ >> 16) & 0xFFFFu) |
                             (32u << 16)),                   // dim1 hi | tile_dim0
                       (int)16u,                             // tile_dim1=16
                       (int)N_,                              // dim0_stride lo32
                       0, 0 };
      const v4i gz = { 0, 0, 0, 0 };
#if __clang_major__ >= 23
      const v8i gz8 = { 0, 0, 0, 0, 0, 0, 0, 0 };
      __builtin_amdgcn_tensor_load_to_lds(g0, g1, gz, gz, gz8, 0);
#else
      __builtin_amdgcn_tensor_load_to_lds(g0, g1, gz, gz, 0);
#endif
    }
#else
#pragma unroll
    for (int q = 0; q < 4; ++q) {                 // fallback: int4 staging
      const int s = lane + 32 * q;
      const int rr = s >> 3, c4 = (s & 7) * 4;
      *(int4*)&sm.Ms[rr][c4] =
          *(const int4*)&mask[((size_t)b * N_ + i0 + rr) * N_ + j0 + c4];
    }
#endif

    {                                             // V tile 32x64 -> Vt[64][32]
      const __bf16* vrow = &vg[(bh + j0 + lane) * HD_];
#pragma unroll
      for (int c8 = 0; c8 < 8; ++c8) {
        const bf16x8 vv = *(const bf16x8*)(vrow + c8 * 8);
#pragma unroll
        for (int e = 0; e < 8; ++e) sm.Vt[c8 * 8 + e][lane] = vv[e];
      }
    }

    v8f S[2];
#pragma unroll
    for (int t = 0; t < 2; ++t) {                 // scores: two 16-col N-tiles
      S[t] = vzero;
#pragma unroll
      for (int c = 0; c < 2; ++c) {
        const __bf16* kp = kg + (bh + j0 + 16 * t + lm) * HD_ + 32 * c + 8 * hi;
        const bf16x8 k0v = *(const bf16x8*)kp;
        const bf16x8 k1v = *(const bf16x8*)(kp + 16);
        v16bf bk;                                  // K^T as B frag
#pragma unroll
        for (int e = 0; e < 8; ++e) { bk[e] = k0v[e]; bk[8 + e] = k1v[e]; }
        S[t] = wmma_bf16(aq[c], bk, S[t]);
      }
    }
    __syncthreads();                              // Vt visible
#ifdef USE_TDM
    __builtin_amdgcn_s_wait_tensorcnt(0);         // Ms DMA complete
#endif

#pragma unroll
    for (int r = 0; r < 8; ++r) {                 // online softmax per row
      const int mr = r + 8 * hi;
      const bool ok0 = sm.Ms[mr][lm] > 0;
      const bool ok1 = sm.Ms[mr][16 + lm] > 0;
      const float s0 = ok0 ? S[0][r] * SCALE_ : -1e30f;
      const float s1 = ok1 ? S[1][r] * SCALE_ : -1e30f;
      float mx = fmaxf(s0, s1);
#pragma unroll
      for (int d = 1; d < 16; d <<= 1) mx = fmaxf(mx, __shfl_xor(mx, d, 32));
      const float mnew = fmaxf(mrow[r], mx);
      const float p0 = ok0 ? __expf(s0 - mnew) : 0.f;
      const float p1 = ok1 ? __expf(s1 - mnew) : 0.f;
      float rs = p0 + p1;
#pragma unroll
      for (int d = 1; d < 16; d <<= 1) rs += __shfl_xor(rs, d, 32);
      const float alpha = __expf(mrow[r] - mnew);
      lrow[r] = lrow[r] * alpha + rs;
      mrow[r] = mnew;
      sm.Ps[mr][lm]      = (__bf16)p0;
      sm.Ps[mr][16 + lm] = (__bf16)p1;
#pragma unroll
      for (int ct = 0; ct < 4; ++ct) acc[ct][r] *= alpha;
    }
    __syncthreads();                              // Ps ready (D->A relayout)

    v16bf ap;                                     // P as A frag (16x32)
#pragma unroll
    for (int e = 0; e < 16; ++e) {
      const int kk = 16 * hi + e;
      ap[e] = sm.Ps[lm][kk];
    }
#pragma unroll
    for (int ct = 0; ct < 4; ++ct) {              // P @ V, contiguous B frag
      v16bf bv;
#pragma unroll
      for (int e = 0; e < 16; ++e) bv[e] = sm.Vt[ct * 16 + lm][16 * hi + e];
      acc[ct] = wmma_bf16(ap, bv, acc[ct]);
    }
    __syncthreads();                              // before next tile overwrite
  }

#pragma unroll
  for (int ct = 0; ct < 4; ++ct)
#pragma unroll
    for (int r = 0; r < 8; ++r) {
      const float o = (lrow[r] > 0.f) ? acc[ct][r] / lrow[r] : 0.f;
      const int row_n = i0 + r + 8 * hi;
      og[((size_t)b * N_ + row_n) * INNER_ + h * HD_ + ct * 16 + lm] = (__bf16)o;
    }
}

// ---------------------------------------------------------------------------
// Kernel 5: output GEMM + bias, f32 result (transposed B staging as above)
// ---------------------------------------------------------------------------
__global__ void __launch_bounds__(128)
gemm_out_kernel(const __bf16* __restrict__ og, const __bf16* __restrict__ wo,
                const float* __restrict__ bias, float* __restrict__ out) {
  __shared__ __bf16 As[64][32];
  __shared__ __bf16 Bt[64][32];
  const int c0 = blockIdx.x * 64;
  const int row0 = blockIdx.y * 64;
  const int tid = threadIdx.x, lane = tid & 31, wave = tid >> 5;
  const int hi = lane >> 4, lm = lane & 15;

  const v8f vzero = {};
  v8f acc[4];
#pragma unroll
  for (int i = 0; i < 4; ++i) acc[i] = vzero;

  for (int k0 = 0; k0 < INNER_; k0 += 32) {
#pragma unroll
    for (int q = 0; q < 2; ++q) {
      const int ch = tid + 128 * q;
      const int r = ch >> 2, c8 = (ch & 3) * 8;
      *(bf16x8*)&As[r][c8] =
          *(const bf16x8*)&og[(size_t)(row0 + r) * INNER_ + k0 + c8];
    }
#pragma unroll
    for (int q = 0; q < 2; ++q) {
      const int ch = tid + 128 * q;
      const int r = ch >> 3, c8 = (ch & 7) * 8;
      const bf16x8 w = *(const bf16x8*)&wo[(size_t)(k0 + r) * D_ + c0 + c8];
#pragma unroll
      for (int e = 0; e < 8; ++e) Bt[c8 + e][r] = w[e];
    }
    if (k0 + 32 < INNER_)
      __builtin_prefetch(&og[(size_t)row0 * INNER_ + k0 + 32], 0, 1);
    __syncthreads();

    v16bf a;
#pragma unroll
    for (int e = 0; e < 16; ++e) {
      const int kk = 16 * hi + e;
      a[e] = As[wave * 16 + lm][kk];
    }
#pragma unroll
    for (int ct = 0; ct < 4; ++ct) {
      v16bf bb;
#pragma unroll
      for (int e = 0; e < 16; ++e) bb[e] = Bt[ct * 16 + lm][16 * hi + e];
      acc[ct] = wmma_bf16(a, bb, acc[ct]);
    }
    __syncthreads();
  }

#pragma unroll
  for (int ct = 0; ct < 4; ++ct)
#pragma unroll
    for (int r = 0; r < 8; ++r) {
      const int grow = row0 + wave * 16 + r + 8 * hi;
      const int col = c0 + ct * 16 + lm;
      out[(size_t)grow * D_ + col] = acc[ct][r] + bias[col];
    }
}

// ---------------------------------------------------------------------------
extern "C" void kernel_launch(void* const* d_in, const int* in_sizes, int n_in,
                              void* d_out, int out_size, void* d_ws,
                              size_t ws_size, hipStream_t stream) {
  (void)in_sizes; (void)n_in; (void)out_size; (void)ws_size;
  const float* x     = (const float*)d_in[0];
  const int*   amask = (const int*)d_in[1];
  const float* gamma = (const float*)d_in[2];
  const float* beta  = (const float*)d_in[3];
  const float* w_qkv = (const float*)d_in[4];
  const float* w_out = (const float*)d_in[5];
  const float* b_out = (const float*)d_in[6];
  float* out = (float*)d_out;

  char* ws = (char*)d_ws;
  __bf16* xn  = (__bf16*)ws; ws += (size_t)ROWS * D_ * 2;          // 8 MiB
  __bf16* wqb = (__bf16*)ws; ws += (size_t)D_ * NC3 * 2;           // 6 MiB
  __bf16* wob = (__bf16*)ws; ws += (size_t)INNER_ * D_ * 2;        // 2 MiB
  __bf16* qg  = (__bf16*)ws; ws += (size_t)B_ * H_ * N_ * HD_ * 2; // 8 MiB
  __bf16* kg  = (__bf16*)ws; ws += (size_t)B_ * H_ * N_ * HD_ * 2; // 8 MiB
  __bf16* vg  = (__bf16*)ws; ws += (size_t)B_ * H_ * N_ * HD_ * 2; // 8 MiB
  __bf16* og  = (__bf16*)ws; ws += (size_t)ROWS * INNER_ * 2;      // 8 MiB

  ln_kernel<<<ROWS, 256, 0, stream>>>(x, gamma, beta, xn);
  cvt_kernel<<<2048, 256, 0, stream>>>(w_qkv, wqb, D_ * NC3);
  cvt_kernel<<<1024, 256, 0, stream>>>(w_out, wob, INNER_ * D_);
  gemm_qkv_kernel<<<dim3(NC3 / 64, ROWS / 64), 128, 0, stream>>>(xn, wqb, qg, kg, vg);
  flash_kernel<<<dim3(N_ / 16, H_, B_), 32, 0, stream>>>(qg, kg, vg, amask, og);
  gemm_out_kernel<<<dim3(D_ / 64, ROWS / 64), 128, 0, stream>>>(og, wob, b_out, out);
}
